// EfficientTransformerBlock_67164698575297
// MI455X (gfx1250) — matrix operand order repaired and checked by reading, not compiled, not run on hardware
//
#include <hip/hip_runtime.h>
#include <hip/hip_bf16.h>
#include <math.h>

typedef _Float16 half_t;
typedef __attribute__((ext_vector_type(8)))  _Float16 v8h;
typedef __attribute__((ext_vector_type(16))) _Float16 v16h;
typedef __attribute__((ext_vector_type(8)))  float    v8f;
typedef __attribute__((ext_vector_type(4)))  float    v4f;

#define DIM   384
#define NHEAD 12
#define HD    32
#define NWIN  64
#define HID   1536

__device__ __forceinline__ v16h cat16(v8h lo, v8h hi) {
  return __builtin_shufflevector(lo, hi, 0,1,2,3,4,5,6,7,8,9,10,11,12,13,14,15);
}

__device__ __forceinline__ v8f wmma16(v16h a, v16h b, v8f c) {
  // D = A(16x32 f16) * B(32x16 f16) + C(16x16 f32)
  return __builtin_amdgcn_wmma_f32_16x16x32_f16(false, a, false, b, (short)0, c, false, false);
}

// A fragment (16x32 f16) from row-major A (contiguous k = 16*(lane/16) + s, as in packed B).
__device__ __forceinline__ v16h load_a_frag(const half_t* __restrict__ A, int ldA, int row0, int k0) {
  int lane = threadIdx.x & 31;
  int m  = row0 + (lane & 15);
  int kb = k0 + ((lane >> 4) << 4);
  const half_t* p = A + (size_t)m * ldA + kb;
  v8h lo = *(const v8h*)p;
  v8h hi = *(const v8h*)(p + 8);
  return cat16(lo, hi);
}

// B fragment from pre-packed weights: one contiguous 32B chunk per lane.
__device__ __forceinline__ v16h load_b_packed(const half_t* __restrict__ Bp, int kTiles, int ntAbs, int kt) {
  int lane = threadIdx.x & 31;
  const half_t* p = Bp + ((((size_t)ntAbs * kTiles + kt) * 32 + lane) << 4);
  v8h lo = *(const v8h*)p;
  v8h hi = *(const v8h*)(p + 8);
  return cat16(lo, hi);
}

// ---------------- weight packing: W(KxN f32) -> B-fragment-major f16 ----------------
__global__ void pack_b_kernel(const float* __restrict__ W, half_t* __restrict__ Bp, int K, int N) {
  int kTiles = K >> 5;
  size_t total = (size_t)K * N;
  size_t tid = (size_t)blockIdx.x * blockDim.x + threadIdx.x;
  if (tid >= total) return;
  int i    = (int)(tid & 15);
  int lane = (int)((tid >> 4) & 31);
  size_t rest = tid >> 9;
  int kt = (int)(rest % kTiles);
  int nt = (int)(rest / kTiles);
  int n = nt * 16 + (lane & 15);
  int k = kt * 32 + ((lane >> 4) << 4) + i;   // B layout: lanes0-15 K=0..15, lanes16-31 K=16..31
  Bp[tid] = (half_t)W[(size_t)k * N + n];
}

// ---- relative position bias, permuted so a lane reads its 4 values as one b128 ----
// biasP[h][q][(key&15)*4 + (key>>4)] = table[relidx(q,key)][h]
__global__ void relbias_kernel(const float* __restrict__ table, float* __restrict__ biasP) {
  int tid = blockIdx.x * blockDim.x + threadIdx.x;
  if (tid >= NHEAD * NWIN * NWIN) return;
  int h = tid >> 12;
  int q = (tid >> 6) & 63;
  int key = tid & 63;
  int r1 = q >> 3, c1 = q & 7, r2 = key >> 3, c2 = key & 7;
  int idx = (r1 - r2 + 7) * 15 + (c1 - c2 + 7);
  biasP[(h << 12) + (q << 6) + ((key & 15) << 2) + (key >> 4)] = table[idx * NHEAD + h];
}

// ---------------- layernorm: f32 in -> f16 out, one wave per token ----------------
__global__ void ln_kernel(const float* __restrict__ x, const float* __restrict__ g,
                          const float* __restrict__ b, half_t* __restrict__ out, int M) {
  int wave = threadIdx.x >> 5, lane = threadIdx.x & 31;
  int tok = blockIdx.x * 8 + wave;
  if (tok >= M) return;
  const float* row = x + (size_t)tok * DIM;
  float v[12]; float s = 0.f, ss = 0.f;
#pragma unroll
  for (int i = 0; i < 12; ++i) { v[i] = row[lane + i * 32]; s += v[i]; ss += v[i] * v[i]; }
#pragma unroll
  for (int m = 16; m >= 1; m >>= 1) { s += __shfl_xor(s, m, 32); ss += __shfl_xor(ss, m, 32); }
  float mu  = s * (1.f / DIM);
  float var = ss * (1.f / DIM) - mu * mu;
  float inv = rsqrtf(var + 1e-5f);
  half_t* orow = out + (size_t)tok * DIM;
#pragma unroll
  for (int i = 0; i < 12; ++i) {
    int c = lane + i * 32;
    orow[c] = (half_t)((v[i] - mu) * inv * g[c] + b[c]);
  }
}

// ---------------- generic WMMA GEMM: one wave computes 16x64 strip ----------------
// Block = 8 waves sharing one N-group over consecutive M-tiles (B frags hit L0).
// EPI 0: out=f16(D+bias); EPI 1: out=f16(gelu(D+bias)); EPI 2: out=f32(D+bias+res)
template <int EPI>
__global__ void gemm_kernel(const half_t* __restrict__ A, const half_t* __restrict__ Bp,
                            const float* __restrict__ bias, const float* __restrict__ res,
                            void* __restrict__ out, int M, int N, int K) {
  int wave = threadIdx.x >> 5, lane = threadIdx.x & 31;
  int mTiles  = M >> 4;
  int nGroups = N >> 6;
  int id = blockIdx.x * 8 + wave;
  if (id >= mTiles * nGroups) return;
  int nG = id / mTiles;          // same for all 8 waves of a block (mTiles % 8 == 0)
  int mT = id % mTiles;
  int kTiles = K >> 5;
  v8f acc[4] = {};
  for (int kt = 0; kt < kTiles; ++kt) {
    v16h a = load_a_frag(A, K, mT * 16, kt * 32);
#pragma unroll
    for (int nt = 0; nt < 4; ++nt) {
      v16h b = load_b_packed(Bp, kTiles, nG * 4 + nt, kt);
      acc[nt] = wmma16(a, b, acc[nt]);
    }
  }
  int mBase = mT * 16 + ((lane >> 4) << 3);
#pragma unroll
  for (int nt = 0; nt < 4; ++nt) {
    int n = nG * 64 + nt * 16 + (lane & 15);
    float bn = bias[n];
#pragma unroll
    for (int r = 0; r < 8; ++r) {
      size_t m = (size_t)(mBase + r);
      float v = acc[nt][r] + bn;
      if (EPI == 1) v = 0.5f * v * (1.0f + erff(v * 0.70710678f));
      if (EPI == 2) {
        ((float*)out)[m * N + n] = v + res[m * N + n];
      } else {
        ((half_t*)out)[m * N + n] = (half_t)v;
      }
    }
  }
}

// ---------------- windowed attention: block = (window, 4 heads), wave = head ----------------
// Unnormalized-exp P in LDS; row sums via WMMA against a ones matrix; single
// per-tile max (softmax-invariant). mt loops kept rolled to bound VGPR pressure.
__global__ void attn_kernel(const half_t* __restrict__ qkv, const float* __restrict__ relBias,
                            half_t* __restrict__ attnOut) {
  __shared__ int    tokIdx[NWIN];
  __shared__ half_t pBuf[4][NWIN][NWIN];  // exp(S - C), per wave
  __shared__ half_t vT[4][HD][NWIN];      // V transposed (dim, token), per wave
  int wave = threadIdx.x >> 5, lane = threadIdx.x & 31;
  int w = blockIdx.x;                      // window
  int h = blockIdx.y * 4 + wave;           // head

  if (threadIdx.x < NWIN) {                // shifted-window token gather indices
    int n = threadIdx.x;
    int bimg = w >> 8;
    int wy = (w >> 4) & 15, wx = w & 15;
    int row = (wy * 8 + (n >> 3) + 4) & 127;
    int col = (wx * 8 + (n & 7) + 4) & 127;
    tokIdx[n] = (bimg << 14) + (row << 7) + col;
  }
  __syncthreads();

  // stage V^T into LDS: 64 tokens x 32 dims, f16
  for (int c = lane; c < 256; c += 32) {
    int t = c >> 2;
    int d0 = (c & 3) << 3;
    v8h val = *(const v8h*)(qkv + (size_t)tokIdx[t] * (3 * DIM) + 2 * DIM + h * HD + d0);
#pragma unroll
    for (int j = 0; j < 8; ++j) vT[wave][d0 + j][t] = val[j];
  }

  // Preload per-lane gather indices for the 4 Q tiles (kills ds->vmem dep chain).
  int tq[4];
#pragma unroll
  for (int t = 0; t < 4; ++t) tq[t] = tokIdx[t * 16 + (lane & 15)];

  // K^T B-fragments stay resident; stream over the 4 query tiles.
  v16h kf[4];
#pragma unroll
  for (int t = 0; t < 4; ++t) {
    int kb = (lane >> 4) << 4;
    const half_t* p = qkv + (size_t)tq[t] * (3 * DIM) + DIM + h * HD + kb;
    kf[t] = cat16(*(const v8h*)p, *(const v8h*)(p + 8));
  }

  const float scale = 0.17677669529663687f;  // 32^-0.5
  int mOff = (lane >> 4) << 3;

#pragma unroll 1   // keep rolled: unrolling would hold 4x S[4] accumulators -> spills
  for (int mt = 0; mt < 4; ++mt) {
    v16h qf;
    {
      int kb = (lane >> 4) << 4;
      const half_t* p = qkv + (size_t)tq[mt] * (3 * DIM) + h * HD + kb;
      qf = cat16(*(const v8h*)p, *(const v8h*)(p + 8));
    }
    v8f S[4];
#pragma unroll
    for (int nt = 0; nt < 4; ++nt) {
      v8f z = {};
      S[nt] = wmma16(qf, kf[nt], z);
    }
    // scale + bias (one b128 per row per lane thanks to permuted bias layout)
    const float* bbase = relBias + ((size_t)h << 12) + ((lane & 15) << 2);
#pragma unroll
    for (int r = 0; r < 8; ++r) {
      int m = mt * 16 + mOff + r;
      v4f bv = *(const v4f*)(bbase + (m << 6));
#pragma unroll
      for (int nt = 0; nt < 4; ++nt) S[nt][r] = S[nt][r] * scale + bv[nt];
    }
    // single max over the whole 16x64 tile: uniform per row => exact softmax
    float mx = -1e30f;
#pragma unroll
    for (int nt = 0; nt < 4; ++nt)
#pragma unroll
      for (int r = 0; r < 8; ++r) mx = fmaxf(mx, S[nt][r]);
#pragma unroll
    for (int msk = 16; msk >= 1; msk >>= 1) mx = fmaxf(mx, __shfl_xor(mx, msk, 32));
    // unnormalized exp -> LDS (row sums recovered via ones-WMMA in the PV phase)
#pragma unroll
    for (int r = 0; r < 8; ++r) {
      int m = mt * 16 + mOff + r;
#pragma unroll
      for (int nt = 0; nt < 4; ++nt)
        pBuf[wave][m][nt * 16 + (lane & 15)] = (half_t)__expf(S[nt][r] - mx);
    }
  }
  __syncthreads();

  // ones fragment for row-sum WMMA
  v16h ones;
#pragma unroll
  for (int i = 0; i < 16; ++i) ones[i] = (half_t)1.0f;

  // O = P @ V, Osum = P @ ones; normalize at the end.
#pragma unroll 1   // keep rolled to bound accumulator pressure
  for (int mt = 0; mt < 4; ++mt) {
    v8f O0 = {}, O1 = {}, Osum = {};
#pragma unroll
    for (int kt = 0; kt < 2; ++kt) {
      int m = mt * 16 + (lane & 15);
      int kb = kt * 32 + ((lane >> 4) << 4);
      v16h pa = cat16(*(const v8h*)&pBuf[wave][m][kb], *(const v8h*)&pBuf[wave][m][kb + 8]);
      int kbv = kt * 32 + ((lane >> 4) << 4);
      int d = lane & 15;
      v16h vb0 = cat16(*(const v8h*)&vT[wave][d][kbv],      *(const v8h*)&vT[wave][d][kbv + 8]);
      v16h vb1 = cat16(*(const v8h*)&vT[wave][d + 16][kbv], *(const v8h*)&vT[wave][d + 16][kbv + 8]);
      O0   = wmma16(pa, vb0, O0);
      O1   = wmma16(pa, vb1, O1);
      Osum = wmma16(pa, ones, Osum);
    }
    // scatter back through token index (fuses window-reverse + unshift)
#pragma unroll
    for (int r = 0; r < 8; ++r) {
      int m = mt * 16 + mOff + r;
      float inv = 1.0f / Osum[r];
      size_t base = (size_t)tokIdx[m] * DIM + h * HD;
      attnOut[base + (lane & 15)]      = (half_t)(O0[r] * inv);
      attnOut[base + 16 + (lane & 15)] = (half_t)(O1[r] * inv);
    }
  }
}

extern "C" void kernel_launch(void* const* d_in, const int* in_sizes, int n_in,
                              void* d_out, int out_size, void* d_ws, size_t ws_size,
                              hipStream_t stream) {
  const float* x         = (const float*)d_in[0];
  const float* qkv_w     = (const float*)d_in[1];
  const float* qkv_b     = (const float*)d_in[2];
  const float* proj_w    = (const float*)d_in[3];
  const float* proj_b    = (const float*)d_in[4];
  const float* rel_table = (const float*)d_in[5];
  const float* n1_g = (const float*)d_in[6];
  const float* n1_b = (const float*)d_in[7];
  const float* n2_g = (const float*)d_in[8];
  const float* n2_b = (const float*)d_in[9];
  const float* w1 = (const float*)d_in[10];
  const float* b1 = (const float*)d_in[11];
  const float* w2 = (const float*)d_in[12];
  const float* b2 = (const float*)d_in[13];

  const int M = in_sizes[0] / DIM;          // tokens (131072 for B=8,H=W=128)
  const int nWindows = M / NWIN;            // 2048

  char* ws = (char*)d_ws;
  size_t off = 0;
  auto alloc = [&](size_t bytes) -> void* {
    void* p = ws + off;
    off = (off + bytes + 255) & ~(size_t)255;
    return p;
  };
  half_t* qkvWp  = (half_t*)alloc((size_t)DIM * 3 * DIM * 2);
  half_t* projWp = (half_t*)alloc((size_t)DIM * DIM * 2);
  half_t* w1p    = (half_t*)alloc((size_t)DIM * HID * 2);
  half_t* w2p    = (half_t*)alloc((size_t)HID * DIM * 2);
  float*  relB   = (float*)alloc((size_t)NHEAD * NWIN * NWIN * 4);
  half_t* hBuf   = (half_t*)alloc((size_t)M * DIM * 2);       // LN1 out, reused for LN2 out
  half_t* qkvBuf = (half_t*)alloc((size_t)M * 3 * DIM * 2);
  half_t* attnB  = (half_t*)alloc((size_t)M * DIM * 2);
  float*  x2Buf  = (float*)alloc((size_t)M * DIM * 4);
  half_t* mBuf   = (half_t*)alloc((size_t)M * HID * 2);

  auto cdiv = [](size_t a, size_t b) { return (int)((a + b - 1) / b); };

  // weight packing + rel-bias table
  pack_b_kernel<<<cdiv((size_t)DIM * 3 * DIM, 256), 256, 0, stream>>>(qkv_w,  qkvWp,  DIM, 3 * DIM);
  pack_b_kernel<<<cdiv((size_t)DIM * DIM,     256), 256, 0, stream>>>(proj_w, projWp, DIM, DIM);
  pack_b_kernel<<<cdiv((size_t)DIM * HID,     256), 256, 0, stream>>>(w1,     w1p,    DIM, HID);
  pack_b_kernel<<<cdiv((size_t)HID * DIM,     256), 256, 0, stream>>>(w2,     w2p,    HID, DIM);
  relbias_kernel<<<cdiv(NHEAD * NWIN * NWIN, 256), 256, 0, stream>>>(rel_table, relB);

  const int mTiles = M / 16;
  auto gemmGrid = [&](int N) { return cdiv((size_t)mTiles * (N / 64), 8); };

  // LN1 -> h (f16)
  ln_kernel<<<cdiv(M, 8), 256, 0, stream>>>(x, n1_g, n1_b, hBuf, M);
  // QKV = h @ qkv_w + qkv_b  (f16)
  gemm_kernel<0><<<gemmGrid(3 * DIM), 256, 0, stream>>>(hBuf, qkvWp, qkv_b, nullptr, qkvBuf, M, 3 * DIM, DIM);
  // windowed attention (shift fused via gather indices)
  attn_kernel<<<dim3(nWindows, NHEAD / 4), 128, 0, stream>>>(qkvBuf, relB, attnB);
  // x2 = x + attn @ proj_w + proj_b  (f32)
  gemm_kernel<2><<<gemmGrid(DIM), 256, 0, stream>>>(attnB, projWp, proj_b, x, x2Buf, M, DIM, DIM);
  // LN2 -> h (f16, reuse)
  ln_kernel<<<cdiv(M, 8), 256, 0, stream>>>(x2Buf, n2_g, n2_b, hBuf, M);
  // m = gelu(h @ w1 + b1)  (f16)
  gemm_kernel<1><<<gemmGrid(HID), 256, 0, stream>>>(hBuf, w1p, b1, nullptr, mBuf, M, HID, DIM);
  // out = x2 + m @ w2 + b2  (f32)
  gemm_kernel<2><<<gemmGrid(DIM), 256, 0, stream>>>(mBuf, w2p, b2, x2Buf, (float*)d_out, M, DIM, HID);
}
